// MultiHeadAttentionLayer_50319836840742
// MI455X (gfx1250) — matrix operand order repaired and checked by reading, not compiled, not run on hardware
//
#include <hip/hip_runtime.h>
#include <hip/hip_bf16.h>
#include <cstdint>

// ---------------------------------------------------------------------------
// MHA for N=2, S=T=2048, E=1024, H=16, HD=64 on gfx1250 (CDNA5, wave32).
// All matmuls on v_wmma_f32_16x16x32_bf16; softmax fused flash-style.
// Weights pre-transposed so ALL WMMA operand loads are contiguous dwords
// (compiler merges them into global_load_b128).
// ---------------------------------------------------------------------------

#define MHA_N  2
#define MHA_S  2048
#define MHA_T  2048
#define MHA_E  1024
#define MHA_H  16
#define MHA_HD 64
#define CHUNK  512          // keys per flash-attention chunk

typedef __attribute__((ext_vector_type(16))) __bf16 v16bf;
typedef __attribute__((ext_vector_type(8)))  float  v8f;

union Frag {                // one WMMA A/B operand: 8 VGPRs = 16 bf16
    uint32_t u[8];
    v16bf    v;
};

// fp32 -> bf16, round-to-nearest-even
static __device__ __forceinline__ uint16_t cvt_bf16(float f) {
    uint32_t x = __builtin_bit_cast(uint32_t, f);
    uint32_t r = x + 0x7fffu + ((x >> 16) & 1u);
    return (uint16_t)(r >> 16);
}

// K-offset of VGPR v for the 16-bit A matrix (16x32), per ISA 7.12.2:
//   VGPR 0..3: K = half*8 + 2v ; VGPR 4..7: K = 16 + half*8 + 2(v-4)
static __device__ __forceinline__ int a_koff(int v, int half) {
    return (v < 4) ? (half * 8 + 2 * v) : (16 + half * 8 + 2 * (v - 4));
}
// K-offset of VGPR v for the 16-bit B matrix (32x16): same K order as A.
static __device__ __forceinline__ int b_koff(int v, int half) {
    return (v < 4) ? (half * 8 + 2 * v) : (16 + half * 8 + 2 * (v - 4));
}

// ---------------------------------------------------------------------------
// Kernel 1a: fp32 -> bf16 (activations, row-major preserved)
// ---------------------------------------------------------------------------
__global__ void mha_cvt_bf16(const float* __restrict__ in,
                             uint16_t* __restrict__ out, size_t n) {
    size_t i = (size_t)blockIdx.x * blockDim.x + threadIdx.x;
    if (i < n) out[i] = cvt_bf16(in[i]);
}

// ---------------------------------------------------------------------------
// Kernel 1b: fp32 W[K][NC] -> bf16 Wt[NC][K]  (transpose + convert)
// ---------------------------------------------------------------------------
__global__ void mha_cvt_bf16_t(const float* __restrict__ in,
                               uint16_t* __restrict__ out, int K, int NC) {
    size_t i = (size_t)blockIdx.x * blockDim.x + threadIdx.x;
    if (i < (size_t)K * NC) {
        int k = (int)(i / NC);
        int n = (int)(i % NC);
        out[(size_t)n * K + k] = cvt_bf16(in[i]);
    }
}

// ---------------------------------------------------------------------------
// Kernel 2/4: bf16 GEMM  C[4096 x 1024] = A[4096 x 1024] * Bt^T + bias
//   Bt is COLUMN-major (Bt[n][k]) -> B fragments load as contiguous dwords.
//   One wave computes a 16x64 strip (4 WMMA tiles sharing one A fragment).
// MODE 0: bf16 row-major out.   MODE 1: fp32 row-major out (final proj).
// MODE 2: bf16 transposed head layout [N,H,HD,T] (for V).
// ---------------------------------------------------------------------------
template <int MODE>
__global__ __launch_bounds__(128)
void mha_gemm_bf16(const uint16_t* __restrict__ A,
                   const uint16_t* __restrict__ Bt,
                   const float*    __restrict__ bias,
                   uint16_t* __restrict__ outBf,
                   float*    __restrict__ outF) {
    constexpr int K  = MHA_E;
    constexpr int NC = MHA_E;

    const int lane = threadIdx.x & 31;
    const int wave = threadIdx.x >> 5;
    const int half = lane >> 4;
    const int lr   = lane & 15;

    const int task = blockIdx.x * 4 + wave;   // 4096 tasks = 256 (M) x 16 (N/64)
    const int tm   = task >> 4;
    const int tn4  = task & 15;

    const size_t arow = (size_t)(tm * 16 + lr) * K;
    int cols[4];
#pragma unroll
    for (int sub = 0; sub < 4; ++sub) cols[sub] = tn4 * 64 + sub * 16 + lr;

    v8f c[4] = {};
    for (int kk = 0; kk < K; kk += 32) {
        Frag a;
#pragma unroll
        for (int v = 0; v < 8; ++v)
            a.u[v] = *(const uint32_t*)(A + arow + kk + a_koff(v, half));
        if (kk + 32 < K)
            __builtin_prefetch(A + arow + kk + 32, 0, 1);   // global_prefetch_b8
#pragma unroll
        for (int sub = 0; sub < 4; ++sub) {
            Frag b;
            const size_t brow = (size_t)cols[sub] * K;
#pragma unroll
            for (int v = 0; v < 8; ++v)
                b.u[v] = *(const uint32_t*)(Bt + brow + kk + b_koff(v, half));
            c[sub] = __builtin_amdgcn_wmma_f32_16x16x32_bf16(
                false, a.v, false, b.v, (short)0, c[sub], false, false);
        }
    }

#pragma unroll
    for (int sub = 0; sub < 4; ++sub) {
        const int col  = cols[sub];
        const float bb = bias[col];
#pragma unroll
        for (int v = 0; v < 8; ++v) {
            const int row = tm * 16 + v + 8 * half;   // C layout: M = v + 8*half
            const float val = c[sub][v] + bb;
            if constexpr (MODE == 0) {
                outBf[(size_t)row * NC + col] = cvt_bf16(val);
            } else if constexpr (MODE == 1) {
                outF[(size_t)row * NC + col] = val;
            } else {                                   // MODE 2: V -> [N,H,HD,T]
                const int nb = row >> 11;              // row / S
                const int s  = row & (MHA_S - 1);
                const int h  = col >> 6;               // col / HD
                const int d  = col & (MHA_HD - 1);
                outBf[(((size_t)nb * MHA_H + h) * MHA_HD + d) * MHA_T + s]
                    = cvt_bf16(val);
            }
        }
    }
}

// ---------------------------------------------------------------------------
// Kernel 3: fused flash attention.
// One block (4 waves) per (n, h, 16-row S block).
//   scores = (Q Kt)/8 -> online softmax in LDS -> O += P V, all via WMMA.
// Q fragments loop-invariant (loaded once). V pre-transposed [N,H,HD,T].
// Softmax scans vectorized: float4 LDS reads, ushort4 bf16 writes.
// ---------------------------------------------------------------------------
__global__ __launch_bounds__(128)
void mha_flash_attn(const uint16_t* __restrict__ Qp,
                    const uint16_t* __restrict__ Kp,
                    const uint16_t* __restrict__ Vt,
                    uint16_t* __restrict__ Ctx) {
    __shared__ float    sc[16][CHUNK + 4];     // raw scores; row stride 2064 B (16B-aligned)
    __shared__ uint16_t pb[16][CHUNK + 8];     // probs bf16; row stride 1040 B (8B-aligned)
    __shared__ float    red[16][8];
    __shared__ float    mrow[16], lrow[16], arow[16];

    const int tid  = threadIdx.x;
    const int lane = tid & 31;
    const int wave = tid >> 5;
    const int half = lane >> 4;
    const int lr   = lane & 15;

    const int sblocks = MHA_S / 16;            // 128
    const int n  = blockIdx.x / (MHA_H * sblocks);
    const int rm = blockIdx.x % (MHA_H * sblocks);
    const int h  = rm / sblocks;
    const int s0 = (rm % sblocks) * 16;

    if (tid < 16) { mrow[tid] = -__builtin_inff(); lrow[tid] = 0.f; }
    __syncthreads();

    // Q fragments for this 16-row block: invariant across chunks & key tiles.
    const size_t qbase = ((size_t)(n * MHA_S + s0 + lr)) * MHA_E + h * MHA_HD;
    Frag aQ[2];
#pragma unroll
    for (int ki = 0; ki < 2; ++ki)
#pragma unroll
        for (int v = 0; v < 8; ++v)
            aQ[ki].u[v] = *(const uint32_t*)(Qp + qbase + ki * 32 + a_koff(v, half));

    v8f o = {};                                // wave's 16x16 of O (cols wave*16..)
    const int d = wave * 16 + lr;              // lane's head-dim column for P*V
    const size_t vbase = (((size_t)n * MHA_H + h) * MHA_HD + d) * MHA_T;

    for (int t0 = 0; t0 < MHA_T; t0 += CHUNK) {
        // ---- phase 1: score tiles (Q 16x64) x (K^T 64x16), 2 WMMA each ----
        for (int j = wave; j < CHUNK / 16; j += 4) {
            const int t = t0 + j * 16 + lr;    // lane's key column
            const size_t kbase = ((size_t)(n * MHA_T + t)) * MHA_E + h * MHA_HD;
            v8f c = {};
#pragma unroll
            for (int ki = 0; ki < 2; ++ki) {
                Frag b;
#pragma unroll
                for (int v = 0; v < 8; ++v)
                    b.u[v] = *(const uint32_t*)(Kp + kbase + ki * 32 + b_koff(v, half));
                c = __builtin_amdgcn_wmma_f32_16x16x32_bf16(
                    false, aQ[ki].v, false, b.v, (short)0, c, false, false);
            }
#pragma unroll
            for (int v = 0; v < 8; ++v)
                sc[v + 8 * half][j * 16 + lr] = c[v] * 0.125f;   // 1/sqrt(64)
        }
        __syncthreads();

        // ---- phase 2: online softmax on the chunk (8 groups x 16 rows) ----
        {
            const int r = tid & 15;            // row
            const int g = tid >> 4;            // group: 64 columns each
            const float4* __restrict__ scv =
                (const float4*)&sc[r][g * 64];             // 16B-aligned
            float4 mx4 = scv[0];
#pragma unroll
            for (int ci = 1; ci < 16; ++ci) {
                const float4 x = scv[ci];
                mx4.x = fmaxf(mx4.x, x.x); mx4.y = fmaxf(mx4.y, x.y);
                mx4.z = fmaxf(mx4.z, x.z); mx4.w = fmaxf(mx4.w, x.w);
            }
            red[r][g] = fmaxf(fmaxf(mx4.x, mx4.y), fmaxf(mx4.z, mx4.w));
            __syncthreads();
            if (g == 0) {
                float cm = red[r][0];
#pragma unroll
                for (int i = 1; i < 8; ++i) cm = fmaxf(cm, red[r][i]);
                const float mOld = mrow[r];
                const float mNew = fmaxf(mOld, cm);
                arow[r] = __expf(mOld - mNew);     // exp(-inf) = 0 on chunk 0
                mrow[r] = mNew;
            }
            __syncthreads();
            const float mNew = mrow[r];
            ushort4* __restrict__ pbv = (ushort4*)&pb[r][g * 64];  // 8B-aligned
            float lsum = 0.f;
#pragma unroll
            for (int ci = 0; ci < 16; ++ci) {
                const float4 x = scv[ci];
                const float e0 = __expf(x.x - mNew);
                const float e1 = __expf(x.y - mNew);
                const float e2 = __expf(x.z - mNew);
                const float e3 = __expf(x.w - mNew);
                ushort4 p;
                p.x = cvt_bf16(e0); p.y = cvt_bf16(e1);
                p.z = cvt_bf16(e2); p.w = cvt_bf16(e3);
                pbv[ci] = p;
                lsum += (e0 + e1) + (e2 + e3);
            }
            red[r][g] = lsum;
            __syncthreads();
            if (g == 0) {
                float s = 0.f;
#pragma unroll
                for (int i = 0; i < 8; ++i) s += red[r][i];
                lrow[r] = lrow[r] * arow[r] + s;
            }
        }
        __syncthreads();

        // ---- phase 3: rescale O, then O += P(16xCHUNK) * V(CHUNKx64) ----
#pragma unroll
        for (int v = 0; v < 8; ++v) o[v] *= arow[v + 8 * half];

#pragma unroll 2
        for (int ks = 0; ks < CHUNK; ks += 32) {
            Frag a, b;
#pragma unroll
            for (int v = 0; v < 8; ++v)
                a.u[v] = *(const uint32_t*)&pb[lr][ks + a_koff(v, half)];
#pragma unroll
            for (int v = 0; v < 8; ++v)
                b.u[v] = *(const uint32_t*)(Vt + vbase + t0 + ks + b_koff(v, half));
            o = __builtin_amdgcn_wmma_f32_16x16x32_bf16(
                false, a.v, false, b.v, (short)0, o, false, false);
        }
        __syncthreads();
    }

    // ---- epilogue: O /= l, write bf16 context [N,S,E] ----
#pragma unroll
    for (int v = 0; v < 8; ++v) {
        const int M = v + 8 * half;
        const float val = o[v] / lrow[M];
        Ctx[((size_t)(n * MHA_S + s0 + M)) * MHA_E + h * MHA_HD + wave * 16 + lr]
            = cvt_bf16(val);
    }
}

// ---------------------------------------------------------------------------
// Host-side launcher
// ---------------------------------------------------------------------------
extern "C" void kernel_launch(void* const* d_in, const int* in_sizes, int n_in,
                              void* d_out, int out_size, void* d_ws, size_t ws_size,
                              hipStream_t stream) {
    const float* query = (const float*)d_in[0];
    const float* key_  = (const float*)d_in[1];
    const float* value = (const float*)d_in[2];
    const float* Wq    = (const float*)d_in[3];
    const float* bq    = (const float*)d_in[4];
    const float* Wk    = (const float*)d_in[5];
    const float* bk    = (const float*)d_in[6];
    const float* Wv    = (const float*)d_in[7];
    const float* bv    = (const float*)d_in[8];
    const float* Wo    = (const float*)d_in[9];
    const float* bo    = (const float*)d_in[10];
    float* out = (float*)d_out;

    const size_t NSE = (size_t)MHA_N * MHA_S * MHA_E;   // 4M elems
    const size_t EE  = (size_t)MHA_E * MHA_E;           // 1M elems

    uint16_t* ws  = (uint16_t*)d_ws;
    uint16_t* qbf = ws;                 // bf16 inputs (row-major)
    uint16_t* kbf = qbf + NSE;
    uint16_t* vbf = kbf + NSE;
    uint16_t* wqt = vbf + NSE;          // bf16 weights, TRANSPOSED [n][k]
    uint16_t* wkt = wqt + EE;
    uint16_t* wvt = wkt + EE;
    uint16_t* wot = wvt + EE;
    uint16_t* Qp  = wot + EE;           // projected Q/K (bf16, [N,S,E])
    uint16_t* Kp  = Qp + NSE;
    uint16_t* Vt  = Kp + NSE;           // projected V, transposed [N,H,HD,T]
    uint16_t* Ctx = Vt + NSE;           // attention context (bf16, [N,S,E])

    {   // activation conversions
        const int blocks = (int)((NSE + 255) / 256);
        mha_cvt_bf16<<<blocks, 256, 0, stream>>>(query, qbf, NSE);
        mha_cvt_bf16<<<blocks, 256, 0, stream>>>(key_,  kbf, NSE);
        mha_cvt_bf16<<<blocks, 256, 0, stream>>>(value, vbf, NSE);
    }
    {   // weight conversions (+transpose)
        const int blocks = (int)((EE + 255) / 256);
        mha_cvt_bf16_t<<<blocks, 256, 0, stream>>>(Wq, wqt, MHA_E, MHA_E);
        mha_cvt_bf16_t<<<blocks, 256, 0, stream>>>(Wk, wkt, MHA_E, MHA_E);
        mha_cvt_bf16_t<<<blocks, 256, 0, stream>>>(Wv, wvt, MHA_E, MHA_E);
        mha_cvt_bf16_t<<<blocks, 256, 0, stream>>>(Wo, wot, MHA_E, MHA_E);
    }

    const int gemmBlocks = (MHA_N * MHA_S / 16) * (MHA_E / 64) / 4;  // 1024

    mha_gemm_bf16<0><<<gemmBlocks, 128, 0, stream>>>(qbf, wqt, bq, Qp, nullptr);
    mha_gemm_bf16<0><<<gemmBlocks, 128, 0, stream>>>(kbf, wkt, bk, Kp, nullptr);
    mha_gemm_bf16<2><<<gemmBlocks, 128, 0, stream>>>(vbf, wvt, bv, Vt, nullptr);

    const int attnBlocks = MHA_N * MHA_H * (MHA_S / 16);     // 4096
    mha_flash_attn<<<attnBlocks, 128, 0, stream>>>(Qp, Kp, Vt, Ctx);

    mha_gemm_bf16<1><<<gemmBlocks, 128, 0, stream>>>(Ctx, wot, bo, nullptr, out);
    (void)in_sizes; (void)n_in; (void)out_size; (void)ws_size;
}
